// PPNet_46780783788032
// MI455X (gfx1250) — matrix operand order repaired and checked by reading, not compiled, not run on hardware
//
#include <hip/hip_runtime.h>
#include <math.h>

// ---------------------------------------------------------------------------
// Async global->LDS copy (CDNA5 GLOBAL_LOAD_ASYNC_TO_LDS_B128, ASYNCcnt).
// Per-lane LDS destination lets us scatter 16B chunks into padded LDS rows.
// Probe-discovered signature: (v4i addrspace(1)*, v4i addrspace(3)*, i32, i32)
// ---------------------------------------------------------------------------
#if __has_builtin(__builtin_amdgcn_global_load_async_to_lds_b128)
#define USE_ASYNC_LDS 1
#endif

typedef int v4i_vs __attribute__((vector_size(16)));   // matches builtin param

__device__ __forceinline__ void cp16_async(void* lds, const void* g) {
#ifdef USE_ASYNC_LDS
  __builtin_amdgcn_global_load_async_to_lds_b128(
      (__attribute__((address_space(1))) v4i_vs*)(void*)g,
      (__attribute__((address_space(3))) v4i_vs*)lds, 0, 0);
#else
  *(uint4*)lds = *(const uint4*)g;              // synchronous fallback
#endif
}
__device__ __forceinline__ void wait_async_copies() {
#ifdef USE_ASYNC_LDS
#if __has_builtin(__builtin_amdgcn_s_wait_asynccnt)
  __builtin_amdgcn_s_wait_asynccnt(0);
#else
  asm volatile("s_wait_asynccnt 0" ::: "memory");
#endif
#endif
}

// ---------------------------------------------------------------------------
// WMMA helpers (wave32): f32 16x16x4 (exact) and bf16 16x16x32 (fast path)
// ---------------------------------------------------------------------------
typedef float  v2f  __attribute__((ext_vector_type(2)));
typedef float  v8f  __attribute__((ext_vector_type(8)));
typedef __bf16 v8bf  __attribute__((ext_vector_type(8)));
typedef __bf16 v16bf __attribute__((ext_vector_type(16)));

__device__ __forceinline__ v8f wmma4(v2f a, v2f b, v8f c) {
  return __builtin_amdgcn_wmma_f32_16x16x4_f32(false, a, false, b, (short)0, c,
                                               false, false);
}
__device__ __forceinline__ v8f wmma_bf16(v16bf a, v16bf b, v8f c) {
  return __builtin_amdgcn_wmma_f32_16x16x32_bf16(false, a, false, b, (short)0,
                                                 c, false, false);
}

// ---- f32 16x16x4 fragments (LDS [M][K] / [K][N], float) -------------------
__device__ __forceinline__ v2f ldsA(const float* A, int stride, int mBase,
                                    int k, int lane) {
  const float* p = A + (mBase + (lane & 15)) * stride + k + ((lane >> 4) << 1);
  v2f r; r.x = p[0]; r.y = p[1];
  return r;
}
__device__ __forceinline__ v2f ldsB(const float* B, int stride, int k,
                                    int nBase, int lane) {
  const float* p = B + (k + ((lane >> 4) << 1)) * stride + nBase + (lane & 15);
  v2f r; r.x = p[0]; r.y = p[stride];
  return r;
}

// ---- bf16 16x16x32 fragments ----------------------------------------------
// A 16x32: lane (h = lane>>4, m = lane&15) holds K = {8h..8h+7, 16+8h..+7}
__device__ __forceinline__ v16bf ldsAbf(const __bf16* A, int stride, int mBase,
                                        int kBase, int lane) {
  const int h = lane >> 4;
  const __bf16* p = A + (mBase + (lane & 15)) * stride + kBase + 8 * h;
  v8bf c0 = *(const v8bf*)p;
  v8bf c1 = *(const v8bf*)(p + 16);
  return __builtin_shufflevector(c0, c1, 0, 1, 2, 3, 4, 5, 6, 7, 8, 9, 10, 11,
                                 12, 13, 14, 15);
}
// B 32x16: lane (h, n) holds K = {8h..8h+7, 16+8h..+7}; B stored transposed [N][K]
__device__ __forceinline__ v16bf ldsBbf(const __bf16* Bt, int stride, int kBase,
                                        int nBase, int lane) {
  const __bf16* p = Bt + (nBase + (lane & 15)) * stride + kBase + 8 * (lane >> 4);
  v8bf c0 = *(const v8bf*)p;
  v8bf c1 = *(const v8bf*)(p + 16);
  return __builtin_shufflevector(c0, c1, 0, 1, 2, 3, 4, 5, 6, 7, 8, 9, 10, 11,
                                 12, 13, 14, 15);
}

__device__ __forceinline__ void split_bf16(float v, __bf16& hi, __bf16& lo) {
  hi = (__bf16)v;
  lo = (__bf16)(v - (float)hi);
}

// ---------------------------------------------------------------------------
// Prep kernel: split prototypes into bf16 hi/lo and compute p2 (exact f32)
// ---------------------------------------------------------------------------
__global__ void k_prep(const float* __restrict__ protos,
                       __bf16* __restrict__ p_hi, __bf16* __restrict__ p_lo,
                       float* __restrict__ p2) {
  int i = blockIdx.x * 256 + threadIdx.x;
  if (i < 2000 * 64) {
    __bf16 hi, lo; split_bf16(protos[i], hi, lo);
    p_hi[i] = hi; p_lo[i] = lo;
  }
  if (i < 2000) {
    float s = 0.f;
    #pragma unroll 8
    for (int k = 0; k < 64; ++k) { float v = protos[i * 64 + k]; s += v * v; }
    p2[i] = s;
  }
}

// ---------------------------------------------------------------------------
// Kernel 1: f = sigmoid(W2 @ relu(W1 @ x + b1) + b2)   (exact f32 WMMA)
// Produces: f pre-split hi/lo bf16, TRANSPOSED [n][l][64]  +  x2[l] = sum f^2
// grid = 32 batches x 8 L-tiles of 64, block = 256 (8 waves)
// ---------------------------------------------------------------------------
__global__ __launch_bounds__(256) void k1_addon(
    const float* __restrict__ x,  const float* __restrict__ W1,
    const float* __restrict__ b1, const float* __restrict__ W2,
    const float* __restrict__ b2, __bf16* __restrict__ f_hi,
    __bf16* __restrict__ f_lo, float* __restrict__ x2_ws) {
  constexpr int CIN = 128, CP = 64, L = 512, LT = 64;
  constexpr int SW1 = CIN + 2, SX = LT + 4, SH = LT + 4, SW2 = CP + 2;
  __shared__ float sW1[CP * SW1];   // weights1 [m][k]
  __shared__ float sX [CIN * SX];   // x tile   [k][l]
  __shared__ float sH [CP * SH];    // h tile   [k][l]  (then reused for f)
  __shared__ float sW2[CP * SW2];   // weights2 [m][k]

  const int n = blockIdx.x >> 3;
  const int lBase = (blockIdx.x & 7) * LT;
  const int tid = threadIdx.x, lane = tid & 31, wave = tid >> 5;

  for (int i = tid; i < CP * CIN; i += 256) sW1[(i >> 7) * SW1 + (i & 127)] = W1[i];
  for (int i = tid; i < CP * CP;  i += 256) sW2[(i >> 6) * SW2 + (i & 63)]  = W2[i];
  for (int i = tid; i < CIN * LT; i += 256) {
    int k = i >> 6, l = i & 63;
    sX[k * SX + l] = x[(n * CIN + k) * L + lBase + l];
  }
  __syncthreads();

  // 16 output tiles of 16x16 (M=64,N=64) -> 2 tiles per wave, same N column.
  const int t0 = wave * 2;
  const int nt = t0 >> 2, mt0 = t0 & 3, mt1 = mt0 + 1;
  const int half = lane >> 4, col = lane & 15;

  // ---- stage A: h = relu(W1 @ x + b1), K = 128 ----
  v8f accA0 = {}; v8f accA1 = {};
  #pragma unroll 4
  for (int k = 0; k < CIN; k += 4) {
    v2f a0 = ldsA(sW1, SW1, mt0 * 16, k, lane);
    v2f a1 = ldsA(sW1, SW1, mt1 * 16, k, lane);
    v2f b  = ldsB(sX, SX, k, nt * 16, lane);
    accA0 = wmma4(a0, b, accA0);
    accA1 = wmma4(a1, b, accA1);
  }
  #pragma unroll
  for (int r = 0; r < 8; ++r) {
    int m0 = mt0 * 16 + r + half * 8;
    int m1 = mt1 * 16 + r + half * 8;
    int l  = nt * 16 + col;
    float v0 = accA0[r] + b1[m0]; sH[m0 * SH + l] = v0 > 0.f ? v0 : 0.f;
    float v1 = accA1[r] + b1[m1]; sH[m1 * SH + l] = v1 > 0.f ? v1 : 0.f;
  }
  __syncthreads();

  // ---- stage B: f = sigmoid(W2 @ h + b2), K = 64 ----
  v8f accB0 = {}; v8f accB1 = {};
  #pragma unroll 4
  for (int k = 0; k < CP; k += 4) {
    v2f a0 = ldsA(sW2, SW2, mt0 * 16, k, lane);
    v2f a1 = ldsA(sW2, SW2, mt1 * 16, k, lane);
    v2f b  = ldsB(sH, SH, k, nt * 16, lane);
    accB0 = wmma4(a0, b, accB0);
    accB1 = wmma4(a1, b, accB1);
  }
  __syncthreads();                       // all reads of sH done -> reuse for f
  #pragma unroll
  for (int r = 0; r < 8; ++r) {
    int m0 = mt0 * 16 + r + half * 8;
    int m1 = mt1 * 16 + r + half * 8;
    int l  = nt * 16 + col;
    float v0 = accB0[r] + b2[m0]; v0 = 1.f / (1.f + __expf(-v0));
    float v1 = accB1[r] + b2[m1]; v1 = 1.f / (1.f + __expf(-v1));
    sH[m0 * SH + l] = v0;
    sH[m1 * SH + l] = v1;
  }
  __syncthreads();

  // x2 = per-position channel sum of f^2 (exact f32)
  if (tid < LT) {
    float s = 0.f;
    #pragma unroll 8
    for (int m = 0; m < CP; ++m) { float v = sH[m * SH + tid]; s += v * v; }
    x2_ws[n * L + lBase + tid] = s;
  }

  // transpose + hi/lo split: f_hi/f_lo[n][l][64] in 16B chunks
  for (int t = tid; t < LT * 8; t += 256) {       // 512 chunk tasks
    int l = t >> 3, c = t & 7;
    v8bf hv, lv;
    #pragma unroll
    for (int j = 0; j < 8; ++j) {
      __bf16 hi, lo; split_bf16(sH[(c * 8 + j) * SH + l], hi, lo);
      hv[j] = hi; lv[j] = lo;
    }
    const int row = (n * L + lBase + l) * CP + c * 8;
    *(v8bf*)(f_hi + row) = hv;
    *(v8bf*)(f_lo + row) = lv;
  }
}

// ---------------------------------------------------------------------------
// Kernel 2 (dominant, 8.4 GFLOP): fused protos@f GEMM + distance + min-over-L
// bf16x3 error-compensated WMMA: xp ~= hi*hi + lo*hi + hi*lo  (f32 accum).
// LDS fills are pure 16B-chunk copies -> GLOBAL_LOAD_ASYNC_TO_LDS_B128,
// scattered into padded (stride-72) bank-conflict-free rows.
// grid = 32 batches x 16 blocks of 128 prototypes, block = 256 (8 waves)
// ---------------------------------------------------------------------------
__global__ __launch_bounds__(256) void k2_proto_min(
    const __bf16* __restrict__ f_hi, const __bf16* __restrict__ f_lo,
    const float* __restrict__ x2_ws, const float* __restrict__ p2_ws,
    const __bf16* __restrict__ p_hi, const __bf16* __restrict__ p_lo,
    float* __restrict__ min_out) {
  constexpr int CP = 64, L = 512, P = 2000, LT = 128, PT = 128;
  constexpr int SB = CP + 8;                    // 72 bf16 = 144 B (16B multiple)
  __shared__ __align__(16) __bf16 sPhi[PT * SB];   // prototypes [p][k]
  __shared__ __align__(16) __bf16 sPlo[PT * SB];
  __shared__ __align__(16) __bf16 sFhi[LT * SB];   // f slab, transposed [l][k]
  __shared__ __align__(16) __bf16 sFlo[LT * SB];
  __shared__ float sX2[LT];
  __shared__ float sP2[PT];
  __shared__ float sMin[PT * 4];                   // [p_local][ngroup]

  const int n = blockIdx.x >> 4;
  const int pBase = (blockIdx.x & 15) * PT;
  const int tid = threadIdx.x, lane = tid & 31, wave = tid >> 5;
  const int mg = wave & 1, ng = wave >> 1;         // 2 M-groups x 4 N-groups
  const int half = lane >> 4, col = lane & 15;

  // prototype tiles: async 16B chunks, row-scattered to stride-72 LDS rows
  for (int t = tid; t < PT * 8; t += 256) {
    int p = t >> 3, c = t & 7;
    int gp = pBase + p; if (gp >= P) gp = P - 1;   // clamp (write is guarded)
    cp16_async(&sPhi[p * SB + c * 8], p_hi + gp * CP + c * 8);
    cp16_async(&sPlo[p * SB + c * 8], p_lo + gp * CP + c * 8);
  }
  if (tid < PT) {
    int gp = pBase + tid; if (gp >= P) gp = P - 1;
    sP2[tid] = p2_ws[gp];
  }

  float rmin[4][8];
  #pragma unroll
  for (int mt = 0; mt < 4; ++mt)
    #pragma unroll
    for (int r = 0; r < 8; ++r) rmin[mt][r] = 3.402823466e38f;

  for (int lt = 0; lt < 4; ++lt) {
    const int lBase = lt * LT;
    __syncthreads();                               // previous slab consumed
    for (int t = tid; t < LT * 8; t += 256) {      // f slab: async chunk copies
      int l = t >> 3, c = t & 7;
      const int grow = (n * L + lBase + l) * CP + c * 8;
      cp16_async(&sFhi[l * SB + c * 8], f_hi + grow);
      cp16_async(&sFlo[l * SB + c * 8], f_lo + grow);
    }
    if (tid < LT) sX2[tid] = x2_ws[n * L + lBase + tid];
    wait_async_copies();                           // ASYNCcnt -> 0
    __syncthreads();

    v8f acc[4][2] = {};
    #pragma unroll
    for (int kb = 0; kb < CP; kb += 32) {          // 2 K-steps x 24 wmma
      v16bf ah[4], al[4], bh[2], bl[2];
      #pragma unroll
      for (int mt = 0; mt < 4; ++mt) {
        ah[mt] = ldsAbf(sPhi, SB, (mg * 4 + mt) * 16, kb, lane);
        al[mt] = ldsAbf(sPlo, SB, (mg * 4 + mt) * 16, kb, lane);
      }
      #pragma unroll
      for (int nt = 0; nt < 2; ++nt) {
        bh[nt] = ldsBbf(sFhi, SB, kb, (ng * 2 + nt) * 16, lane);
        bl[nt] = ldsBbf(sFlo, SB, kb, (ng * 2 + nt) * 16, lane);
      }
      #pragma unroll
      for (int mt = 0; mt < 4; ++mt)
        #pragma unroll
        for (int nt = 0; nt < 2; ++nt) {
          acc[mt][nt] = wmma_bf16(ah[mt], bh[nt], acc[mt][nt]);
          acc[mt][nt] = wmma_bf16(al[mt], bh[nt], acc[mt][nt]);
          acc[mt][nt] = wmma_bf16(ah[mt], bl[nt], acc[mt][nt]);
        }
    }
    // distance + running min in accumulator layout
    #pragma unroll
    for (int mt = 0; mt < 4; ++mt) {
      #pragma unroll
      for (int r = 0; r < 8; ++r) {
        const int pl = mg * 64 + mt * 16 + r + half * 8;
        const float pp = sP2[pl];
        #pragma unroll
        for (int nt = 0; nt < 2; ++nt) {
          const int ll = ng * 32 + nt * 16 + col;
          float d = sX2[ll] - 2.f * acc[mt][nt][r] + pp;
          d = d > 0.f ? d : 0.f;
          rmin[mt][r] = fminf(rmin[mt][r], d);
        }
      }
    }
  }

  // reduce across the 16 lanes holding one prototype row (xor 1,2,4,8 stays
  // inside each 16-lane half -> rows r and r+8 reduce independently)
  #pragma unroll
  for (int mt = 0; mt < 4; ++mt)
    #pragma unroll
    for (int r = 0; r < 8; ++r) {
      float v = rmin[mt][r];
      v = fminf(v, __shfl_xor(v, 1, 32));
      v = fminf(v, __shfl_xor(v, 2, 32));
      v = fminf(v, __shfl_xor(v, 4, 32));
      v = fminf(v, __shfl_xor(v, 8, 32));
      if (col == 0)
        sMin[(mg * 64 + mt * 16 + r + half * 8) * 4 + ng] = v;
    }
  __syncthreads();
  if (tid < PT) {
    int gp = pBase + tid;
    if (gp < P) {
      float m = fminf(fminf(sMin[tid * 4 + 0], sMin[tid * 4 + 1]),
                      fminf(sMin[tid * 4 + 2], sMin[tid * 4 + 3]));
      min_out[n * P + gp] = m;
    }
  }
}

// ---------------------------------------------------------------------------
// Kernel 4: activations + final linear.  One block per batch element.
// ---------------------------------------------------------------------------
__global__ __launch_bounds__(256) void k_logits(
    const float* __restrict__ min_d, const float* __restrict__ last_w,
    float* __restrict__ logits) {
  constexpr int P = 2000, NC = 200;
  __shared__ float sAct[P];
  const int n = blockIdx.x, tid = threadIdx.x;
  for (int p = tid; p < P; p += 256) {
    float m = min_d[n * P + p];
    sAct[p] = logf((m + 1.0f) / (m + 1e-4f));
  }
  __syncthreads();
  for (int c = tid; c < NC; c += 256) {
    float s = 0.f;
    const float* w = last_w + c * P;
    for (int p = 0; p < P; ++p) s += sAct[p] * w[p];
    logits[n * NC + c] = s;
  }
}

// ---------------------------------------------------------------------------
// launch: out = [logits 32*200 | min_distances 32*2000]
// workspace (bytes): f_hi 2Mi | f_lo 2Mi | p_hi 250K | p_lo 250K | x2 64Ki | p2
// ---------------------------------------------------------------------------
extern "C" void kernel_launch(void* const* d_in, const int* in_sizes, int n_in,
                              void* d_out, int out_size, void* d_ws,
                              size_t ws_size, hipStream_t stream) {
  const float* x      = (const float*)d_in[0];
  const float* W1     = (const float*)d_in[1];
  const float* b1     = (const float*)d_in[2];
  const float* W2     = (const float*)d_in[3];
  const float* b2     = (const float*)d_in[4];
  const float* protos = (const float*)d_in[5];
  const float* last_w = (const float*)d_in[6];

  float* out    = (float*)d_out;
  float* logits = out;                  // 32*200
  float* min_d  = out + 32 * 200;       // 32*2000

  char* ws = (char*)d_ws;
  __bf16* f_hi = (__bf16*)ws;                               // 32*512*64 bf16
  __bf16* f_lo = (__bf16*)(ws + 2097152);
  __bf16* p_hi = (__bf16*)(ws + 4194304);                   // 2000*64 bf16
  __bf16* p_lo = (__bf16*)(ws + 4194304 + 256000);
  float*  x2_ws = (float*)(ws + 4194304 + 512000);          // 32*512 f32
  float*  p2_ws = (float*)(ws + 4194304 + 512000 + 65536);  // 2000 f32

  k_prep      <<<500,     256, 0, stream>>>(protos, p_hi, p_lo, p2_ws);
  k1_addon    <<<32 * 8,  256, 0, stream>>>(x, W1, b1, W2, b2, f_hi, f_lo, x2_ws);
  k2_proto_min<<<32 * 16, 256, 0, stream>>>(f_hi, f_lo, x2_ws, p2_ws, p_hi,
                                            p_lo, min_d);
  k_logits    <<<32,      256, 0, stream>>>(min_d, last_w, logits);
}
